// MultiHeadAttention_37563783970974
// MI455X (gfx1250) — matrix operand order repaired and checked by reading, not compiled, not run on hardware
//
#include <hip/hip_runtime.h>
#include <hip/hip_bf16.h>
#include <math.h>

// ---------------------------------------------------------------------------
// MHA forward for MI455X (gfx1250), bf16 WMMA compute / fp32 accumulate.
// B=2, S=2048, D=1024, H=16, DK=64.
// B-side tiles staged block-cooperatively with GLOBAL_LOAD_ASYNC_TO_LDS_B128
// (ASYNCcnt), double-buffered.
// ---------------------------------------------------------------------------

#define Bsz 2
#define Ssz 2048
#define Dsz 1024
#define Hsz 16
#define DKsz 64

typedef __bf16 bf16_t;
typedef __attribute__((ext_vector_type(16))) __bf16 v16bf;
typedef __attribute__((ext_vector_type(4)))  __bf16 v4bf;
typedef __attribute__((ext_vector_type(8)))  float  v8f;

union Frag16 { uint4 q[2]; v16bf v; };

// Load a 16-element bf16 fragment as two 16-byte chunks (p0[0..7], p1[0..7]).
__device__ __forceinline__ v16bf load_frag16(const bf16_t* p0, const bf16_t* p1) {
    Frag16 f;
    f.q[0] = *(const uint4*)p0;
    f.q[1] = *(const uint4*)p1;
    return f.v;
}

__device__ __forceinline__ v8f wmma_bf16(v16bf a, v16bf b, v8f c) {
    // D(f32 16x16) = A(bf16 16x32) * B(bf16 32x16) + C
    return __builtin_amdgcn_wmma_f32_16x16x32_bf16(
        /*neg_a=*/false, a, /*neg_b=*/false, b,
        /*c_mod=*/(short)0, c, /*reuse_a=*/false, /*reuse_b=*/false);
}

__device__ __forceinline__ v8f zero_v8f() {
    v8f z = {0.f,0.f,0.f,0.f,0.f,0.f,0.f,0.f};
    return z;
}

// Async copy of one 16-byte chunk from global (saddr + voff) into LDS[ldsoff].
// GVS mode: saddr = uniform 64-bit base, voff = per-lane 32-bit byte offset.
__device__ __forceinline__ void async_b128_to_lds(unsigned int ldsoff,
                                                  unsigned int voff,
                                                  unsigned long long base) {
    asm volatile("global_load_async_to_lds_b128 %0, %1, %2"
                 :: "v"(ldsoff), "v"(voff), "s"(base) : "memory");
}

__device__ __forceinline__ void wait_async0() {
    asm volatile("s_wait_asynccnt 0x0" ::: "memory");
}

__device__ __forceinline__ unsigned int lds_addr32(const void* p) {
    // ROCm generic shared addresses: low 32 bits carry the LDS byte offset.
    return (unsigned int)(unsigned long long)(uintptr_t)p;
}

// ---------------------------------------------------------------------------
// fp32 -> bf16 pack (x matrix), 4 elements/thread
// ---------------------------------------------------------------------------
__global__ void pack_bf16_kernel(const float* __restrict__ in, bf16_t* __restrict__ out, int n4) {
    int i = blockIdx.x * blockDim.x + threadIdx.x;
    if (i >= n4) return;
    float4 f = ((const float4*)in)[i];
    v4bf o;
    o.x = (bf16_t)f.x; o.y = (bf16_t)f.y; o.z = (bf16_t)f.z; o.w = (bf16_t)f.w;
    *(v4bf*)(out + (size_t)i * 4) = o;
}

// ---------------------------------------------------------------------------
// Transpose-pack weight: W[K][N] fp32 -> Wt[N][K] bf16 (LDS tiled)
// ---------------------------------------------------------------------------
__global__ void transpose_pack_kernel(const float* __restrict__ W, bf16_t* __restrict__ Wt,
                                      int K, int N) {
    __shared__ float tile[32][33];
    int kb = blockIdx.x * 32;
    int nb = blockIdx.y * 32;
#pragma unroll
    for (int i = 0; i < 4; ++i) {
        int k = kb + threadIdx.y + i * 8;
        int n = nb + threadIdx.x;
        tile[threadIdx.y + i * 8][threadIdx.x] = W[(size_t)k * N + n];
    }
    __syncthreads();
#pragma unroll
    for (int i = 0; i < 4; ++i) {
        int n = nb + threadIdx.y + i * 8;
        int k = kb + threadIdx.x;
        Wt[(size_t)n * K + k] = (bf16_t)tile[threadIdx.x][threadIdx.y + i * 8];
    }
}

// ---------------------------------------------------------------------------
// GEMM: C[M,N] = relu(A[M,K](bf16) * Wt[N,K]^T(bf16) + bias)
// Block: 256 thr (8 waves). Wave tile: 16 rows x 64 cols. K-step 32.
// Weight K-slab (64 x 32 = 4KB) staged via async-to-LDS, double-buffered:
// one b128 per thread per tile, so the 8 waves share one copy instead of
// each loading it from global.
// MODE 0: store bf16 as Q/K layout  [(b*H+h)*S + s][dk]      (col = h*64+dk)
// MODE 1: store bf16 as V^T layout  [(b*H+h)*64 + dk][s]
// MODE 2: store fp32 row-major      [row][col]
// ---------------------------------------------------------------------------
template<int MODE>
__global__ void gemm_bf16_kernel(const bf16_t* __restrict__ A,
                                 const bf16_t* __restrict__ Wt,
                                 const float* __restrict__ bias,
                                 void* __restrict__ outp,
                                 int M, int N, int K) {
    __shared__ bf16_t blds[2][64][32];   // [buf][n within tile][k within slab]

    const int tid  = threadIdx.x;
    const int lane = tid & 31;
    const int w    = tid >> 5;
    const int r0   = blockIdx.y * 128 + w * 16;
    const int n0   = blockIdx.x * 64;
    const int lh   = lane >> 4;    // lane half
    const int ll   = lane & 15;

    // async staging pattern: thread -> (row = tid/4, k-chunk = tid%4)
    const int brow = tid >> 2;
    const int bseg = tid & 3;
    const unsigned int bvoff = (unsigned int)(brow * (K * 2) + bseg * 16);

    v8f acc[4];
#pragma unroll
    for (int t = 0; t < 4; ++t) acc[t] = zero_v8f();

    // A fragment row pointer for this lane (lane = M row, half selects K phase)
    const bf16_t* arow = A + (size_t)(r0 + ll) * K + lh * 8;

    // prologue: stage first weight slab
    {
        unsigned long long base = (unsigned long long)(uintptr_t)(Wt + (size_t)n0 * K);
        async_b128_to_lds(lds_addr32(&blds[0][brow][bseg * 8]), bvoff, base);
    }

    for (int kk = 0; kk < K; kk += 32) {
        const int cur = (kk >> 5) & 1;
        wait_async0();
        __syncthreads();
        if (kk + 32 < K) {
            unsigned long long base =
                (unsigned long long)(uintptr_t)(Wt + (size_t)n0 * K + (kk + 32));
            async_b128_to_lds(lds_addr32(&blds[cur ^ 1][brow][bseg * 8]), bvoff, base);
        }

        __builtin_prefetch(arow + kk + 64, 0, 0);             // global_prefetch_b8
        v16bf af = load_frag16(arow + kk, arow + kk + 16);    // A: {k..k+7, k+16..k+23}
#pragma unroll
        for (int t = 0; t < 4; ++t) {
            const bf16_t* brp = &blds[cur][t * 16 + ll][lh * 8];
            v16bf bf = load_frag16(brp, brp + 16);            // B frag from LDS
            acc[t] = wmma_bf16(af, bf, acc[t]);
        }
    }

#pragma unroll
    for (int t = 0; t < 4; ++t) {
        const int col = n0 + t * 16 + ll;
        const float bia = bias[col];
#pragma unroll
        for (int r = 0; r < 8; ++r) {
            const int row = r0 + r + 8 * lh;                  // C layout: M = r + 8*half
            float v = fmaxf(acc[t][r] + bia, 0.0f);           // bias + ReLU
            if (MODE == 2) {
                ((float*)outp)[(size_t)row * N + col] = v;
            } else {
                const int b  = row >> 11;                     // /S
                const int s  = row & (Ssz - 1);
                const int h  = col >> 6;                      // /DK
                const int dk = col & (DKsz - 1);
                const int bh = b * Hsz + h;
                if (MODE == 0)
                    ((bf16_t*)outp)[((size_t)bh * Ssz + s) * DKsz + dk] = (bf16_t)v;
                else
                    ((bf16_t*)outp)[((size_t)bh * DKsz + dk) * Ssz + s] = (bf16_t)v;
            }
        }
    }
}

// ---------------------------------------------------------------------------
// Flash attention: per wave 16 query rows, stream 32 keys/iter, online softmax.
// Q,K: [B*H, S, 64] bf16.  Vt: [B*H, 64, S] bf16.  mask: [B, S] int (1 = masked).
// ctx out: [B*S, H*64] bf16.
// K tile (32x64, contiguous 4KB) and V^T tile (64x32) staged block-wide via
// async-to-LDS (double-buffered) so 8 waves share one copy.
// Reference semantics: score = (q.kT)*dk^-0.5, masked keys replaced by 1e-9
// BEFORE softmax (not -inf).
// ---------------------------------------------------------------------------
__global__ void flash_attn_kernel(const bf16_t* __restrict__ Q,
                                  const bf16_t* __restrict__ Km,
                                  const bf16_t* __restrict__ Vt,
                                  const int* __restrict__ mask,
                                  bf16_t* __restrict__ ctx) {
    __shared__ bf16_t klds[2][32][64];   // [buf][key][dk]
    __shared__ bf16_t vlds[2][64][32];   // [buf][dk][key]
    __shared__ bf16_t plds[8][16][32];   // per-wave P staging for C->A relayout

    const int tid  = threadIdx.x;
    const int lane = tid & 31;
    const int w    = tid >> 5;
    const int lh   = lane >> 4;
    const int ll   = lane & 15;
    const int bh   = blockIdx.y;
    const int b    = bh >> 4;          // / H
    const int h    = bh & (Hsz - 1);
    const int qr0  = blockIdx.x * 128 + w * 16;

    // async staging patterns
    const unsigned int kvoff = (unsigned int)(tid * 16);                  // K: 4KB contiguous
    const unsigned int vvoff = (unsigned int)((tid >> 2) * (Ssz * 2) + (tid & 3) * 16);

    // Q fragments (16 rows x 64, two K=32 slabs), resident all kernel.
    const bf16_t* qbase = Q + ((size_t)bh * Ssz + (qr0 + ll)) * DKsz;
    v16bf qa[2];
#pragma unroll
    for (int kf = 0; kf < 2; ++kf)
        qa[kf] = load_frag16(qbase + kf * 32 + lh * 8,
                             qbase + kf * 32 + lh * 8 + 16);

    float mrow[8], lrow[8];
    v8f o[4];
#pragma unroll
    for (int r = 0; r < 8; ++r) { mrow[r] = -INFINITY; lrow[r] = 0.f; }
#pragma unroll
    for (int nf = 0; nf < 4; ++nf) o[nf] = zero_v8f();

    const float scale = 0.125f;   // DK^-0.5

    // prologue: stage first K/V tile
    {
        unsigned long long kbase =
            (unsigned long long)(uintptr_t)(Km + ((size_t)bh * Ssz) * DKsz);
        async_b128_to_lds(lds_addr32(&klds[0][0][0]) + kvoff, kvoff, kbase);
        unsigned long long vbase =
            (unsigned long long)(uintptr_t)(Vt + (size_t)bh * DKsz * Ssz);
        async_b128_to_lds(lds_addr32(&vlds[0][tid >> 2][(tid & 3) * 8]), vvoff, vbase);
    }

    for (int j0 = 0; j0 < Ssz; j0 += 32) {
        const int cur = (j0 >> 5) & 1;
        wait_async0();
        __syncthreads();
        if (j0 + 32 < Ssz) {
            unsigned long long kbase =
                (unsigned long long)(uintptr_t)(Km + ((size_t)bh * Ssz + (j0 + 32)) * DKsz);
            async_b128_to_lds(lds_addr32(&klds[cur ^ 1][0][0]) + kvoff, kvoff, kbase);
            unsigned long long vbase =
                (unsigned long long)(uintptr_t)(Vt + (size_t)bh * DKsz * Ssz + (j0 + 32));
            async_b128_to_lds(lds_addr32(&vlds[cur ^ 1][tid >> 2][(tid & 3) * 8]), vvoff, vbase);
        }

        // ---- S = Q @ K^T  (16 x 32, two 16x16 C frags), K tile from LDS ----
        v8f sfr[2] = { zero_v8f(), zero_v8f() };
        int  mv[2];
#pragma unroll
        for (int t = 0; t < 2; ++t) {
            mv[t] = mask[b * Ssz + j0 + t * 16 + ll];
            const bf16_t* kb = &klds[cur][t * 16 + ll][0];
#pragma unroll
            for (int kf = 0; kf < 2; ++kf) {
                v16bf bf = load_frag16(kb + kf * 32 + lh * 8,
                                       kb + kf * 32 + lh * 8 + 16);
                sfr[t] = wmma_bf16(qa[kf], bf, sfr[t]);
            }
        }

        // ---- scale + masked_fill(1e-9) ----
        float p[2][8];
#pragma unroll
        for (int t = 0; t < 2; ++t)
#pragma unroll
            for (int r = 0; r < 8; ++r)
                p[t][r] = mv[t] ? 1e-9f : sfr[t][r] * scale;

        // ---- online softmax: row max over 32 keys (16-lane column groups) ----
        float rmax[8];
#pragma unroll
        for (int r = 0; r < 8; ++r) rmax[r] = fmaxf(p[0][r], p[1][r]);
#pragma unroll
        for (int off = 1; off < 16; off <<= 1)
#pragma unroll
            for (int r = 0; r < 8; ++r)
                rmax[r] = fmaxf(rmax[r], __shfl_xor(rmax[r], off, 32));

        float alpha[8];
#pragma unroll
        for (int r = 0; r < 8; ++r) {
            float mn = fmaxf(mrow[r], rmax[r]);
            alpha[r] = __expf(mrow[r] - mn);
            mrow[r]  = mn;
        }

        float rsum[8];
#pragma unroll
        for (int r = 0; r < 8; ++r) rsum[r] = 0.f;
#pragma unroll
        for (int t = 0; t < 2; ++t)
#pragma unroll
            for (int r = 0; r < 8; ++r) {
                p[t][r] = __expf(p[t][r] - mrow[r]);
                rsum[r] += p[t][r];
            }
#pragma unroll
        for (int off = 1; off < 16; off <<= 1)
#pragma unroll
            for (int r = 0; r < 8; ++r)
                rsum[r] += __shfl_xor(rsum[r], off, 32);
#pragma unroll
        for (int r = 0; r < 8; ++r) lrow[r] = lrow[r] * alpha[r] + rsum[r];

        // rescale running O
#pragma unroll
        for (int nf = 0; nf < 4; ++nf)
#pragma unroll
            for (int r = 0; r < 8; ++r) o[nf][r] *= alpha[r];

        // ---- P: C-layout -> A-layout via per-wave LDS ----
#pragma unroll
        for (int t = 0; t < 2; ++t)
#pragma unroll
            for (int r = 0; r < 8; ++r)
                plds[w][r + 8 * lh][t * 16 + ll] = (bf16_t)p[t][r];
        asm volatile("s_wait_dscnt 0x0" ::: "memory");   // same-wave LDS RAW
        const bf16_t* pl = &plds[w][ll][lh * 8];
        v16bf pa = load_frag16(pl, pl + 16);

        // ---- O += P @ V  (K-dim = 32 keys, N = 64 dk), V tile from LDS ----
#pragma unroll
        for (int nf = 0; nf < 4; ++nf) {
            const bf16_t* vb = &vlds[cur][nf * 16 + ll][lh * 8];
            v16bf vf = load_frag16(vb, vb + 16);
            o[nf] = wmma_bf16(pa, vf, o[nf]);
        }
    }

    // ---- normalize and store ctx [(b*S+s)][h*64+dk] ----
#pragma unroll
    for (int r = 0; r < 8; ++r) {
        const float inv = 1.0f / lrow[r];
        const int srow = qr0 + r + 8 * lh;
#pragma unroll
        for (int nf = 0; nf < 4; ++nf) {
            const int dk = nf * 16 + ll;
            ctx[((size_t)(b * Ssz + srow)) * (Hsz * DKsz) + h * DKsz + dk] =
                (bf16_t)(o[nf][r] * inv);
        }
    }
}

// ---------------------------------------------------------------------------
// Launch
// ---------------------------------------------------------------------------
extern "C" void kernel_launch(void* const* d_in, const int* in_sizes, int n_in,
                              void* d_out, int out_size, void* d_ws, size_t ws_size,
                              hipStream_t stream) {
    (void)in_sizes; (void)n_in; (void)out_size; (void)ws_size;

    const float* x    = (const float*)d_in[0];
    const int*   mask = (const int*)d_in[1];
    const float* Wq   = (const float*)d_in[2];
    const float* bq   = (const float*)d_in[3];
    const float* Wk   = (const float*)d_in[4];
    const float* bk   = (const float*)d_in[5];
    const float* Wv   = (const float*)d_in[6];
    const float* bv   = (const float*)d_in[7];
    const float* Wo   = (const float*)d_in[8];
    const float* bo   = (const float*)d_in[9];
    float* out = (float*)d_out;

    char* ws = (char*)d_ws;
    const size_t MB = 1024 * 1024;
    bf16_t* xbf  = (bf16_t*)(ws);                 //  8 MB: x bf16 [4096,1024]
    bf16_t* wqt  = (bf16_t*)(ws +  8 * MB);       //  2 MB: Wq^T bf16 [1024,1024]
    bf16_t* wkt  = (bf16_t*)(ws + 10 * MB);
    bf16_t* wvt  = (bf16_t*)(ws + 12 * MB);
    bf16_t* wot  = (bf16_t*)(ws + 14 * MB);
    bf16_t* qws  = (bf16_t*)(ws + 16 * MB);       //  8 MB: Q [B*H, S, 64]
    bf16_t* kws  = (bf16_t*)(ws + 24 * MB);       //  8 MB: K [B*H, S, 64]
    bf16_t* vtws = (bf16_t*)(ws + 32 * MB);       //  8 MB: V^T [B*H, 64, S]
    bf16_t* ctx  = (bf16_t*)(ws + 40 * MB);       //  8 MB: ctx [B*S, 1024]

    const int M = Bsz * Ssz;        // 4096
    const int N = Hsz * DKsz;       // 1024
    const int K = Dsz;              // 1024

    // 1) pack x -> bf16
    pack_bf16_kernel<<<(M * K / 4 + 255) / 256, 256, 0, stream>>>(x, xbf, M * K / 4);

    // 2) transpose-pack weights -> [N,K] bf16
    dim3 tb(32, 8), tg(K / 32, N / 32);
    transpose_pack_kernel<<<tg, tb, 0, stream>>>(Wq, wqt, K, N);
    transpose_pack_kernel<<<tg, tb, 0, stream>>>(Wk, wkt, K, N);
    transpose_pack_kernel<<<tg, tb, 0, stream>>>(Wv, wvt, K, N);
    transpose_pack_kernel<<<tg, tb, 0, stream>>>(Wo, wot, K, N);

    // 3) QKV projections (relu fused), remapped layouts
    dim3 gg(N / 64, M / 128);
    gemm_bf16_kernel<0><<<gg, 256, 0, stream>>>(xbf, wqt, bq, qws,  M, N, K);
    gemm_bf16_kernel<0><<<gg, 256, 0, stream>>>(xbf, wkt, bk, kws,  M, N, K);
    gemm_bf16_kernel<1><<<gg, 256, 0, stream>>>(xbf, wvt, bv, vtws, M, N, K);

    // 4) flash attention
    dim3 ag(Ssz / 128, Bsz * Hsz);
    flash_attn_kernel<<<ag, 256, 0, stream>>>(qws, kws, vtws, mask, ctx);

    // 5) output projection (relu fused), fp32 out
    gemm_bf16_kernel<2><<<gg, 256, 0, stream>>>(ctx, wot, bo, (void*)out, M, N, K);
}
